// MoELayer_39986145526201
// MI455X (gfx1250) — matrix operand order repaired and checked by reading, not compiled, not run on hardware
//
#include <hip/hip_runtime.h>
#include <hip/hip_bf16.h>

// MoE top-2 (B=4,S=2048,D=1024,O=1024,E=8,K=2), f16 WMMA grouped GEMM.
// Pipeline: zero -> convert W (f32->f16, transposed) -> convert x (f32->f16)
//           -> router (top-2 softmax) -> scatter by expert -> bias init
//           -> grouped GEMM (128x128x64 tiles, double-buffered LDS, WMMA).

constexpr int ND = 1024, NO = 1024, NE = 8;
constexpr int NTOK = 4 * 2048;        // 8192 tokens
constexpr int CAP  = NTOK;            // per-expert gather capacity
constexpr int BM = 128, BN = 128, BK = 64;
constexpr int LDK = BK + 8;           // 72 halves (144B): 16B-aligned, bank-spread
constexpr int NKI = ND / BK;          // 16 k-iterations

typedef __attribute__((ext_vector_type(16))) _Float16     v16h;
typedef __attribute__((ext_vector_type(8)))  float        v8f;
typedef __attribute__((ext_vector_type(4)))  unsigned int v4u;

union H8  { _Float16 h[8]; v4u u; };
union F16 { v4u u[2]; v16h v; };

__global__ void k_zero(int* c) {
    if (threadIdx.x < NE) c[threadIdx.x] = 0;
}

// expert_w [E][D][O] f32  ->  wt [E][O][D] f16 (transposed: K contiguous per out-col)
__global__ void k_convert_wt(const float* __restrict__ ew, _Float16* __restrict__ wt) {
    __shared__ _Float16 tile[32][33];
    const int e  = blockIdx.z;
    const int d0 = blockIdx.y * 32;
    const int o0 = blockIdx.x * 32;
    const float* src = ew + (size_t)e * ND * NO;
    for (int i = threadIdx.x; i < 1024; i += 256) {
        int dl = i >> 5, ol = i & 31;
        tile[dl][ol] = (_Float16)src[(size_t)(d0 + dl) * NO + (o0 + ol)];
    }
    __syncthreads();
    _Float16* dst = wt + (size_t)e * NO * ND;
    for (int i = threadIdx.x; i < 1024; i += 256) {
        int ol = i >> 5, dl = i & 31;
        dst[(size_t)(o0 + ol) * ND + (d0 + dl)] = tile[dl][ol];
    }
}

// x f32 -> xh f16, 8 elements per thread
__global__ void k_convert_x(const float* __restrict__ x, _Float16* __restrict__ xh) {
    const size_t i = ((size_t)blockIdx.x * blockDim.x + threadIdx.x) * 8;
    if (i >= (size_t)NTOK * ND) return;
    const float4* p = (const float4*)(x + i);
    float4 f0 = p[0], f1 = p[1];
    H8 h;
    h.h[0] = (_Float16)f0.x; h.h[1] = (_Float16)f0.y;
    h.h[2] = (_Float16)f0.z; h.h[3] = (_Float16)f0.w;
    h.h[4] = (_Float16)f1.x; h.h[5] = (_Float16)f1.y;
    h.h[6] = (_Float16)f1.z; h.h[7] = (_Float16)f1.w;
    *(v4u*)(xh + i) = h.u;
}

// one wave per token: logits (f32 exact), top-2, softmax over the two kept logits
__global__ void k_router(const float* __restrict__ x, const float* __restrict__ gw,
                         const float* __restrict__ gb,
                         int* __restrict__ idx, float* __restrict__ wgt) {
    const int tok  = (blockIdx.x * blockDim.x + threadIdx.x) >> 5;
    const int lane = threadIdx.x & 31;
    if (tok >= NTOK) return;
    const float* xr = x + (size_t)tok * ND;
    float acc[NE];
#pragma unroll
    for (int e = 0; e < NE; ++e) acc[e] = 0.f;
    for (int j = lane; j < ND; j += 32) {
        float xv = xr[j];
#pragma unroll
        for (int e = 0; e < NE; ++e) acc[e] = fmaf(xv, gw[e * ND + j], acc[e]);
    }
#pragma unroll
    for (int o = 16; o > 0; o >>= 1) {
#pragma unroll
        for (int e = 0; e < NE; ++e) acc[e] += __shfl_xor(acc[e], o, 32);
    }
    if (lane == 0) {
        float l[NE];
#pragma unroll
        for (int e = 0; e < NE; ++e) l[e] = acc[e] + gb[e];
        int i0 = 0; float m0 = l[0];
#pragma unroll
        for (int e = 1; e < NE; ++e) if (l[e] > m0) { m0 = l[e]; i0 = e; }
        int i1 = (i0 == 0) ? 1 : 0; float m1 = l[i1];
#pragma unroll
        for (int e = 0; e < NE; ++e) if (e != i0 && l[e] > m1) { m1 = l[e]; i1 = e; }
        float t  = __expf(m1 - m0);
        float w0 = 1.f / (1.f + t);
        idx[2 * tok]     = i0;  idx[2 * tok + 1] = i1;
        wgt[2 * tok]     = w0;  wgt[2 * tok + 1] = 1.f - w0;
    }
}

__global__ void k_scatter(const int* __restrict__ idx, const float* __restrict__ wgt,
                          int* cursor, int* __restrict__ toks, float* __restrict__ tw) {
    const int t = blockIdx.x * blockDim.x + threadIdx.x;
    if (t >= 2 * NTOK) return;
    const int e   = idx[t];
    const int pos = atomicAdd(&cursor[e], 1);
    toks[e * CAP + pos] = t >> 1;
    tw[e * CAP + pos]   = wgt[t];
}

// out[n][o] = w0*b[e0][o] + w1*b[e1][o]  (bias folded; full overwrite every call)
__global__ void k_init_out(const int* __restrict__ idx, const float* __restrict__ wgt,
                           const float* __restrict__ eb, float* __restrict__ out) {
    const int t = blockIdx.x * blockDim.x + threadIdx.x;
    if (t >= NTOK * NO) return;
    const int n = t >> 10, o = t & (NO - 1);
    const int e0 = idx[2 * n], e1 = idx[2 * n + 1];
    out[t] = wgt[2 * n] * eb[e0 * NO + o] + wgt[2 * n + 1] * eb[e1 * NO + o];
}

// grouped GEMM: gathered token rows (f16) x W_e^T (f16), f32 acc, WMMA 16x16x32.
// 128x128 macro tile, BK=64, double-buffered LDS, register-pipelined global loads.
__global__ void __launch_bounds__(256)
k_moe_gemm(const _Float16* __restrict__ xh, const _Float16* __restrict__ wt,
           const int* __restrict__ toks, const float* __restrict__ tw,
           const int* __restrict__ cursor, float* __restrict__ out) {
    const int e      = blockIdx.z;
    const int cnt    = min(cursor[e], CAP);
    const int m_base = blockIdx.y * BM;
    if (m_base >= cnt) return;
    const int n_base = blockIdx.x * BN;
    const int rows   = min(BM, cnt - m_base);
    const int lb     = e * CAP + m_base;

    __shared__ __align__(16) _Float16 As[2][BM * LDK];
    __shared__ __align__(16) _Float16 Bs[2][BN * LDK];

    const int tid = threadIdx.x;
    const int r   = tid >> 1;               // 0..127 staging row
    const int kh  = (tid & 1) << 5;         // 0 or 32 halves within k-slab
    const bool aok = (r < rows);
    const _Float16* asrc = nullptr;
    if (aok) asrc = xh + (size_t)toks[lb + r] * ND + kh;
    const _Float16* bsrc = wt + ((size_t)e * NO + (size_t)(n_base + r)) * ND + kh;

    const int lane = tid & 31;
    const int wv = tid >> 5;
    const int wm = wv >> 1, wn = wv & 1;    // 4x2 wave grid: 32x64 per wave
    const int lr = lane & 15, lh = lane >> 4;

    v8f acc[2][4] = {};
    v4u ra[4], rb[4];

    auto gload = [&](int k0) {
        if (aok) {
            const v4u* p = (const v4u*)(asrc + k0);
            ra[0] = p[0]; ra[1] = p[1]; ra[2] = p[2]; ra[3] = p[3];
        } else {
            ra[0] = ra[1] = ra[2] = ra[3] = v4u{0, 0, 0, 0};
        }
        const v4u* q = (const v4u*)(bsrc + k0);
        rb[0] = q[0]; rb[1] = q[1]; rb[2] = q[2]; rb[3] = q[3];
    };
    auto lstore = [&](int b) {
        v4u* pa = (v4u*)(&As[b][r * LDK + kh]);
        pa[0] = ra[0]; pa[1] = ra[1]; pa[2] = ra[2]; pa[3] = ra[3];
        v4u* pb = (v4u*)(&Bs[b][r * LDK + kh]);
        pb[0] = rb[0]; pb[1] = rb[1]; pb[2] = rb[2]; pb[3] = rb[3];
    };

    gload(0);
    lstore(0);
    __syncthreads();

    for (int k = 0; k < NKI; ++k) {
        const int cur = k & 1;
        if (k + 1 < NKI) gload((k + 1) * BK);   // overlap with WMMA below

        // two 32-wide k-subtiles per LDS slab
#pragma unroll
        for (int ks = 0; ks < 2; ++ks) {
            v16h af[2], bf[4];
#pragma unroll
            for (int mi = 0; mi < 2; ++mi) {
                F16 f;
                const _Float16* p =
                    &As[cur][(wm * 32 + mi * 16 + lr) * LDK + ks * 32 + lh * 8];
                f.u[0] = *(const v4u*)p;          // K = khalf..khalf+7
                f.u[1] = *(const v4u*)(p + 16);   // K = 16+khalf..
                af[mi] = f.v;
            }
#pragma unroll
            for (int ni = 0; ni < 4; ++ni) {
                F16 f;
                const _Float16* p =
                    &Bs[cur][(wn * 64 + ni * 16 + lr) * LDK + ks * 32 + lh * 8];
                f.u[0] = *(const v4u*)p;          // K = khalf..khalf+7
                f.u[1] = *(const v4u*)(p + 16);   // K = 16+khalf..
                bf[ni] = f.v;
            }
#pragma unroll
            for (int mi = 0; mi < 2; ++mi)
#pragma unroll
                for (int ni = 0; ni < 4; ++ni)
                    acc[mi][ni] = __builtin_amdgcn_wmma_f32_16x16x32_f16(
                        false, af[mi], false, bf[ni], (short)0, acc[mi][ni],
                        false, false);
        }

        if (k + 1 < NKI) lstore(cur ^ 1);
        __syncthreads();
    }

    // ---- epilogue: scale by gate weight, accumulate into token rows ----
#pragma unroll
    for (int mi = 0; mi < 2; ++mi) {
#pragma unroll
        for (int j = 0; j < 8; ++j) {
            const int rloc = wm * 32 + mi * 16 + lh * 8 + j;  // C: M = j + 8*(lane>=16)
            if (rloc < rows) {
                const int token = toks[lb + rloc];
                const float w   = tw[lb + rloc];
                float* orow = out + (size_t)token * NO + n_base + wn * 64 + lr;
#pragma unroll
                for (int ni = 0; ni < 4; ++ni) {
                    __hip_atomic_fetch_add(orow + ni * 16, w * acc[mi][ni][j],
                                           __ATOMIC_RELAXED, __HIP_MEMORY_SCOPE_AGENT);
                }
            }
        }
    }
}

extern "C" void kernel_launch(void* const* d_in, const int* in_sizes, int n_in,
                              void* d_out, int out_size, void* d_ws, size_t ws_size,
                              hipStream_t stream) {
    const float* x  = (const float*)d_in[0];
    const float* gw = (const float*)d_in[1];
    const float* gb = (const float*)d_in[2];
    const float* ew = (const float*)d_in[3];
    const float* eb = (const float*)d_in[4];
    float* out = (float*)d_out;

    char* ws = (char*)d_ws;
    size_t off = 0;
    auto alloc = [&](size_t bytes) -> void* {
        off = (off + 255) & ~(size_t)255;
        void* p = ws + off;
        off += bytes;
        return p;
    };
    int*       cursor = (int*)alloc(NE * sizeof(int));
    int*       idx    = (int*)alloc((size_t)2 * NTOK * sizeof(int));
    float*     wgt    = (float*)alloc((size_t)2 * NTOK * sizeof(float));
    int*       toks   = (int*)alloc((size_t)NE * CAP * sizeof(int));
    float*     tw     = (float*)alloc((size_t)NE * CAP * sizeof(float));
    _Float16*  wt     = (_Float16*)alloc((size_t)NE * NO * ND * sizeof(_Float16));
    _Float16*  xh     = (_Float16*)alloc((size_t)NTOK * ND * sizeof(_Float16));
    (void)ws_size; (void)in_sizes; (void)n_in; (void)out_size;

    k_zero<<<1, 32, 0, stream>>>(cursor);
    k_convert_wt<<<dim3(NO / 32, ND / 32, NE), 256, 0, stream>>>(ew, wt);
    k_convert_x<<<(NTOK * ND / 8) / 256, 256, 0, stream>>>(x, xh);
    k_router<<<NTOK / 8, 256, 0, stream>>>(x, gw, gb, idx, wgt);
    k_scatter<<<(2 * NTOK) / 256, 256, 0, stream>>>(idx, wgt, cursor, toks, tw);
    k_init_out<<<(NTOK * NO) / 256, 256, 0, stream>>>(idx, wgt, eb, out);
    k_moe_gemm<<<dim3(NO / BN, CAP / BM, NE), 256, 0, stream>>>(xh, wt, toks, tw, cursor, out);
}
